// RNNModel_84284438217107
// MI455X (gfx1250) — compile-verified, matrix operand order checked by reading, not run on hardware
//
#include <hip/hip_runtime.h>
#include <math.h>

#define SEQ   1024
#define BATCH 128
#define IN    256
#define HID   512

typedef __attribute__((ext_vector_type(16))) __bf16 v16bf;
typedef __attribute__((ext_vector_type(8)))  float  v8f;

// ---------------------------------------------------------------------------
// Swizzled fragment layout (CDNA5 ISA 7.12.2, wave32):
// fragment element j of lane l maps to K = (j<8?0:16) + (l>=16?8:0) + (j&7)
// A-tile 16x32: lane = M row (l&15), K split by lane half
// B-tile 32x16: lane = N col (l&15), same K split
// A fragment stored contiguously: frag_base + l*16 + j  -> one 32B load/lane.
// ---------------------------------------------------------------------------

// A fragment 16x32 from row-major f32 source (convert to bf16) — phase 1 only.
__device__ inline v16bf load_A_f32(const float* __restrict__ src, int ld, int lane) {
    v16bf a;
    const int m  = lane & 15;
    const int kb = (lane >> 4) << 3;         // 0 or 8
    const float* p = src + (size_t)m * ld + kb;
#pragma unroll
    for (int j = 0; j < 16; j += 2) {
        const int k = ((j & 8) << 1) + (j & 7);   // pairs contiguous in k
        float2 f = *(const float2*)(p + k);
        a[j]     = (__bf16)f.x;
        a[j + 1] = (__bf16)f.y;
    }
    return a;
}

// ---------------------------------------------------------------------------
// Weight prep: transpose + convert + swizzle into per-lane fragment order.
// Tile index = kt * (HID/16) + nt ; element = tile*512 + lane*16 + j.
// ---------------------------------------------------------------------------
__global__ void prep_weights(const float* __restrict__ w_ih, const float* __restrict__ w_hh,
                             __bf16* __restrict__ wihS, __bf16* __restrict__ whhS) {
    const int idx = blockIdx.x * blockDim.x + threadIdx.x;   // 0 .. HID*HID-1
    const int j    = idx & 15;
    const int lane = (idx >> 4) & 31;
    const int tile = idx >> 9;
    const int nt = tile & 31;                 // HID/16 == 32 column tiles (both)
    const int kt = tile >> 5;
    const int n = nt * 16 + (lane & 15);
    const int k = kt * 32 + ((j & 8) << 1) + ((lane >> 4) << 3) + (j & 7);
    if (idx < IN * HID)                        // B[k][n] = w_ih[n][k]
        wihS[idx] = (__bf16)w_ih[n * IN + k];
    if (idx < HID * HID)                       // B[k][n] = w_hh[n][k]
        whhS[idx] = (__bf16)w_hh[n * HID + k];
}

// ---------------------------------------------------------------------------
// Phase 1: xw = x @ W_ih^T  -> d_out (standard layout).
// Block = 16-row strip, 8 waves; wave w covers 64 cols (4 tiles), K-loop of 8.
// ---------------------------------------------------------------------------
__global__ void __launch_bounds__(256) rnn_xw(const float* __restrict__ x,
                                              const __bf16* __restrict__ wihS,
                                              float* __restrict__ out) {
    const int lane = threadIdx.x & 31;
    const int wave = threadIdx.x >> 5;
    const size_t row0 = (size_t)blockIdx.x * 16;
    const int nt0 = wave * 4;                  // first of 4 column tiles
    const int n0  = nt0 * 16;

    v8f c0 = {}, c1 = {}, c2 = {}, c3 = {};
#pragma unroll
    for (int kt = 0; kt < IN / 32; ++kt) {
        v16bf a = load_A_f32(x + row0 * IN + kt * 32, IN, lane);
        const __bf16* tb = wihS + ((size_t)(kt * 32 + nt0) * 32 + lane) * 16;
        v16bf b0 = *(const v16bf*)(tb);
        v16bf b1 = *(const v16bf*)(tb + 512);
        v16bf b2 = *(const v16bf*)(tb + 1024);
        v16bf b3 = *(const v16bf*)(tb + 1536);
        c0 = __builtin_amdgcn_wmma_f32_16x16x32_bf16(false, a, false, b0, (short)0, c0, false, false);
        c1 = __builtin_amdgcn_wmma_f32_16x16x32_bf16(false, a, false, b1, (short)0, c1, false, false);
        c2 = __builtin_amdgcn_wmma_f32_16x16x32_bf16(false, a, false, b2, (short)0, c2, false, false);
        c3 = __builtin_amdgcn_wmma_f32_16x16x32_bf16(false, a, false, b3, (short)0, c3, false, false);
    }
    const int n  = lane & 15;
    const int mb = (lane >> 4) << 3;
#pragma unroll
    for (int r = 0; r < 8; ++r) {
        float* dst = out + (row0 + mb + r) * HID;
        dst[n0 +      n] = c0[r];
        dst[n0 + 16 + n] = c1[r];
        dst[n0 + 32 + n] = c2[r];
        dst[n0 + 48 + n] = c3[r];
    }
}

// ---------------------------------------------------------------------------
// Phase 2: h_t = tanh(xw_t + h_{t-1} @ W_hh^T), in place on d_out.
// 32 blocks x 8 waves = 256 waves; wave owns one 16x16 tile (8 m-strips x 32 n-tiles).
// W_hh fragments (16 x v16bf = 128 VGPRs) preloaded once, reused all 1024 steps.
// h kept double-buffered in A-fragment-swizzled bf16 layout.
// ---------------------------------------------------------------------------
__global__ void __launch_bounds__(256) rnn_recurrent(float* __restrict__ out,
                                                     const __bf16* __restrict__ whhS,
                                                     __bf16* __restrict__ hswz0,
                                                     __bf16* __restrict__ hswz1,
                                                     unsigned* __restrict__ counter) {
    const int lane  = threadIdx.x & 31;
    const int wave  = threadIdx.x >> 5;
    const int ms    = blockIdx.x & 7;                    // batch strip (16 rows)
    const int m0    = ms * 16;
    const int ntile = (blockIdx.x >> 3) * 8 + wave;      // 0..31
    const int n0    = ntile * 16;
    const unsigned nBlocks = gridDim.x;

    // Preload all 16 B fragments for this column tile (invariant over t).
    v16bf bfrag[16];
#pragma unroll
    for (int kt = 0; kt < 16; ++kt)
        bfrag[kt] = *(const v16bf*)(whhS + ((size_t)(kt * 32 + ntile) * 32 + lane) * 16);

    const int n  = lane & 15;
    const int mb = (lane >> 4) << 3;

    // Precompute swizzled h-store indices for this thread's 8 C elements.
    int sidx[8];
    {
        const int nglob  = n0 + n;
        const int klocal = nglob & 31;
        const int kt     = nglob >> 5;
        const int jj     = (klocal & 7) | (((klocal >> 4) & 1) << 3);
        const int laneA_hi = ((klocal >> 3) & 1) << 4;
#pragma unroll
        for (int r = 0; r < 8; ++r) {
            const int laneA = (mb + r) | laneA_hi;
            sidx[r] = ((ms * 16 + kt) * 32 + laneA) * 16 + jj;
        }
    }

    for (int t = 0; t < SEQ; ++t) {
        const __bf16* hp = (t & 1) ? hswz1 : hswz0;
        __bf16*       hn = (t & 1) ? hswz0 : hswz1;
        float* xw_t = out + (size_t)t * BATCH * HID;

        // C <- xw tile
        v8f acc;
#pragma unroll
        for (int r = 0; r < 8; ++r)
            acc[r] = xw_t[(size_t)(m0 + mb + r) * HID + n0 + n];

        // acc += h_prev(16x512) @ W_hh^T(512x16) — 16 chained WMMAs
#pragma unroll
        for (int kt = 0; kt < 16; ++kt) {
            v16bf a = *(const v16bf*)(hp + (size_t)(ms * 16 + kt) * 512 + lane * 16);
            acc = __builtin_amdgcn_wmma_f32_16x16x32_bf16(false, a, false, bfrag[kt],
                                                          (short)0, acc, false, false);
        }

        // tanh + stores: f32 output in place, swizzled bf16 next-h, final h_n tail
#pragma unroll
        for (int r = 0; r < 8; ++r) {
            const float v = tanhf(acc[r]);
            const size_t off = (size_t)(m0 + mb + r) * HID + n0 + n;
            xw_t[off] = v;
            hn[sidx[r]] = (__bf16)v;
            if (t == SEQ - 1)
                out[(size_t)SEQ * BATCH * HID + off] = v;
        }

        // ---- device-wide barrier (release/acquire on monotonic counter) ----
        __threadfence();
        __syncthreads();
        if (threadIdx.x == 0) {
            __hip_atomic_fetch_add(counter, 1u, __ATOMIC_ACQ_REL, __HIP_MEMORY_SCOPE_AGENT);
            const unsigned target = nBlocks * (unsigned)(t + 1);
            while (__hip_atomic_load(counter, __ATOMIC_ACQUIRE, __HIP_MEMORY_SCOPE_AGENT) < target)
                __builtin_amdgcn_s_sleep(1);
        }
        __syncthreads();
        __threadfence();
    }
}

// ---------------------------------------------------------------------------
extern "C" void kernel_launch(void* const* d_in, const int* in_sizes, int n_in,
                              void* d_out, int out_size, void* d_ws, size_t ws_size,
                              hipStream_t stream) {
    const float* x    = (const float*)d_in[0];
    const float* w_ih = (const float*)d_in[1];
    const float* w_hh = (const float*)d_in[2];
    float* out = (float*)d_out;

    // Workspace layout (~1.05 MB):
    //   [0]      counter (zeroed)
    //   [256]    h_swz0  bf16 BATCH*HID (zeroed: h0 = 0)
    //   +128KB   h_swz1  bf16 BATCH*HID
    //   +128KB   wihS    bf16 IN*HID   (swizzled)
    //   +256KB   whhS    bf16 HID*HID  (swizzled)
    char* ws = (char*)d_ws;
    const size_t BH = (size_t)BATCH * HID;          // 65536
    unsigned* counter = (unsigned*)ws;
    __bf16* h0   = (__bf16*)(ws + 256);
    __bf16* h1   = (__bf16*)(ws + 256 + BH * 2);
    __bf16* wihS = (__bf16*)(ws + 256 + BH * 4);
    __bf16* whhS = (__bf16*)(ws + 256 + BH * 4 + (size_t)IN * HID * 2);

    hipMemsetAsync(d_ws, 0, 256 + BH * 2, stream);

    prep_weights<<<(HID * HID + 255) / 256, 256, 0, stream>>>(w_ih, w_hh, wihS, whhS);

    rnn_xw<<<(SEQ * BATCH) / 16, 256, 0, stream>>>(x, wihS, out);

    rnn_recurrent<<<32, 256, 0, stream>>>(out, whhS, h0, h1, counter);
}
